// SlidingWindowAttn_41394894799619
// MI455X (gfx1250) — matrix operand order repaired and checked by reading, not compiled, not run on hardware
//
#include <hip/hip_runtime.h>
#include <math.h>

typedef __attribute__((ext_vector_type(16))) _Float16 v16h;
typedef __attribute__((ext_vector_type(8)))  float    v8f;

#define BB   8
#define HWP  4096   // 64*64
#define NOUT0 2097152   // 8*64*4096
#define NKV   4194304   // 8*64*2*4096

static __device__ __forceinline__ v16h vz16() {
    v16h r;
#pragma unroll
    for (int j = 0; j < 16; ++j) r[j] = (_Float16)0;
    return r;
}

// A operand, row-major A[m][k], lda = K stride. Tile at (m0, k0), K-span 32.
// lane<16: halves0-15 = K(k0..k0+15)
// lane>=16: halves0-15 = K(k0+16..k0+31)
static __device__ __forceinline__ v16h wmma_a_rm(const _Float16* A, int lda, int m0, int k0, int lane) {
    int m = m0 + (lane & 15);
    const _Float16* p = A + (size_t)m * lda + k0 + ((lane >> 4) << 4);
    v16h r;
#pragma unroll
    for (int j = 0; j < 8; ++j) r[j] = p[j];
#pragma unroll
    for (int j = 0; j < 8; ++j) r[8 + j] = p[8 + j];
    return r;
}

// B operand from K-major (transposed) BT[n][k], ldb = K stride.
// lane holds column n = n0+(lane&15), K = k0 + 16*(lane>>4) + j (16 contiguous halves).
static __device__ __forceinline__ v16h wmma_b_km(const _Float16* BT, int ldb, int n0, int k0, int lane) {
    int n = n0 + (lane & 15);
    const _Float16* p = BT + (size_t)n * ldb + k0 + ((lane >> 4) << 4);
    v16h r;
#pragma unroll
    for (int j = 0; j < 16; ++j) r[j] = p[j];
    return r;
}

// ---------------- prep kernels ----------------

// Convert all weight matrices to f16 (conv weights reordered to [O][tap*64+ci]).
__global__ void k_prep_w(const float* Wqkv, const float* Wf1, const float* Wf2, const float* Wproj,
                         _Float16* WqkvH, _Float16* Wf1H, _Float16* Wf2H, _Float16* WprojH) {
    int i = blockIdx.x * blockDim.x + threadIdx.x;
    if (i < 12288) { WqkvH[i] = (_Float16)Wqkv[i]; return; }
    i -= 12288;
    if (i < 36864) {
        int o = i / 576, r = i % 576, tap = r >> 6, ci = r & 63;
        Wf1H[i] = (_Float16)Wf1[(o * 64 + ci) * 9 + tap];
        return;
    }
    i -= 36864;
    if (i < 18432) {
        int o = i / 576, r = i % 576, tap = r >> 6, ci = r & 63;
        Wf2H[i] = (o < 25) ? (_Float16)Wf2[(o * 64 + ci) * 9 + tap] : (_Float16)0;
        return;
    }
    i -= 18432;
    if (i < 4096) WprojH[i] = (_Float16)Wproj[i];
}

// x (b,64,4096) f32 -> xT[b][pixel][ch] f16
__global__ void k_x_f16(const float* x, _Float16* xT) {
    int idx = blockIdx.x * blockDim.x + threadIdx.x;
    if (idx >= BB * HWP * 64) return;
    int c = idx & 63, p = (idx >> 6) & 4095, b = idx >> 18;
    xT[idx] = (_Float16)x[((size_t)b * 64 + c) * HWP + p];
}

// ---------------- qkv = dw3x3( Wqkv @ x ) ----------------

__global__ void k_gemm_qkv(const _Float16* W, const _Float16* xT, float* out) {
    int wave = (blockIdx.x * blockDim.x + threadIdx.x) >> 5;
    int lane = threadIdx.x & 31;
    int nt = wave % 256, mt = (wave / 256) % 12, b = wave / (256 * 12);
    const _Float16* BT = xT + (size_t)b * HWP * 64;
    v8f acc = {};
#pragma unroll
    for (int k0 = 0; k0 < 64; k0 += 32) {
        v16h a = wmma_a_rm(W, 64, mt * 16, k0, lane);
        v16h bm = wmma_b_km(BT, 64, nt * 16, k0, lane);
        acc = __builtin_amdgcn_wmma_f32_16x16x32_f16(false, a, false, bm, (short)0, acc, false, false);
    }
    float* O = out + (size_t)b * 192 * HWP;
    int hi = lane >> 4, nn = nt * 16 + (lane & 15);
#pragma unroll
    for (int v = 0; v < 8; ++v)
        O[(size_t)(mt * 16 + v + 8 * hi) * HWP + nn] = acc[v];
}

__global__ void k_depthwise(const float* qkv0, const float* Wdw, float* qkv) {
    int idx = blockIdx.x * blockDim.x + threadIdx.x;
    if (idx >= BB * 192 * HWP) return;
    int p = idx & 4095, c = (idx >> 12) % 192, b = idx / (192 * HWP);
    int y = p >> 6, x = p & 63;
    const float* src = qkv0 + ((size_t)b * 192 + c) * HWP;
    const float* w = Wdw + c * 9;
    float acc = 0.f;
#pragma unroll
    for (int di = 0; di < 3; ++di)
#pragma unroll
        for (int dj = 0; dj < 3; ++dj) {
            int yy = y + di - 1, xx = x + dj - 1;
            if (yy >= 0 && yy < 64 && xx >= 0 && xx < 64) acc += src[yy * 64 + xx] * w[di * 3 + dj];
        }
    qkv[idx] = acc;
}

// k_ = [k_cached, k], v_ = [v_cached, k]  (faithful: v_ uses k) -> d_out regions
__global__ void k_concat(const float* kc, const float* vc, const float* qkv, float* outk, float* outv) {
    int idx = blockIdx.x * blockDim.x + threadIdx.x;
    if (idx >= BB * 64 * HWP) return;
    int p = idx & 4095, c = (idx >> 12) & 63, b = idx >> 18;
    float kval = qkv[((size_t)b * 192 + 64 + c) * HWP + p];
    size_t base = ((size_t)(b * 64 + c)) * 2 * HWP;
    outk[base + p] = kc[idx];
    outk[base + HWP + p] = kval;
    outv[base + p] = vc[idx];
    outv[base + HWP + p] = kval;
}

// frames -> yT[f][pixel][ch] f16 ; f = tensor*16 + b*2 + t (tensor0=k_, tensor1=v_)
__global__ void k_frames_f16(const float* kreg, const float* vreg, _Float16* yT) {
    int idx = blockIdx.x * blockDim.x + threadIdx.x;
    if (idx >= 32 * HWP * 64) return;
    int c = idx & 63, p = (idx >> 6) & 4095, f = idx >> 18;
    int tensor = f >> 4, bt = f & 15, b = bt >> 1, t = bt & 1;
    const float* src = tensor ? vreg : kreg;
    yT[idx] = (_Float16)src[(((size_t)b * 64 + c) * 2 + t) * HWP + p];
}

// ---------------- dynamic filtering convs (implicit GEMM, K = 9*64) ----------------

__global__ void k_conv1(const _Float16* Wf1, const _Float16* yT, _Float16* fT) {
    int wave = (blockIdx.x * blockDim.x + threadIdx.x) >> 5;
    int lane = threadIdx.x & 31;
    int nt = wave % 256, mt = (wave / 256) & 3, f = wave / (256 * 4);
    const _Float16* Y = yT + (size_t)f * HWP * 64;
    int n = nt * 16 + (lane & 15);
    int py = n >> 6, px = n & 63, hi = lane >> 4;
    v8f acc = {};
    for (int tap = 0; tap < 9; ++tap) {
        int yy = py + tap / 3 - 1, xx = px + tap % 3 - 1;
        bool ok = (yy >= 0 && yy < 64 && xx >= 0 && xx < 64);
#pragma unroll
        for (int kc = 0; kc < 64; kc += 32) {
            v16h a = wmma_a_rm(Wf1, 576, mt * 16, tap * 64 + kc, lane);
            v16h bm = vz16();
            if (ok) {
                const _Float16* p = Y + (size_t)(yy * 64 + xx) * 64 + kc + (hi << 4);
#pragma unroll
                for (int j = 0; j < 16; ++j) bm[j] = p[j];
            }
            acc = __builtin_amdgcn_wmma_f32_16x16x32_f16(false, a, false, bm, (short)0, acc, false, false);
        }
    }
    _Float16* Od = fT + ((size_t)f * HWP + n) * 64 + mt * 16 + 8 * hi;
#pragma unroll
    for (int v = 0; v < 8; ++v) {
        float t = acc[v];
        Od[v] = (_Float16)(t >= 0.f ? t : 0.2f * t);  // LeakyReLU(0.2)
    }
}

__global__ void k_conv2(const _Float16* Wf2, const _Float16* fT, float* ker) {
    int wave = (blockIdx.x * blockDim.x + threadIdx.x) >> 5;
    int lane = threadIdx.x & 31;
    int nt = wave % 256, mt = (wave / 256) & 1, f = wave / (256 * 2);
    const _Float16* Y = fT + (size_t)f * HWP * 64;
    int n = nt * 16 + (lane & 15);
    int py = n >> 6, px = n & 63, hi = lane >> 4;
    v8f acc = {};
    for (int tap = 0; tap < 9; ++tap) {
        int yy = py + tap / 3 - 1, xx = px + tap % 3 - 1;
        bool ok = (yy >= 0 && yy < 64 && xx >= 0 && xx < 64);
#pragma unroll
        for (int kc = 0; kc < 64; kc += 32) {
            v16h a = wmma_a_rm(Wf2, 576, mt * 16, tap * 64 + kc, lane);
            v16h bm = vz16();
            if (ok) {
                const _Float16* p = Y + (size_t)(yy * 64 + xx) * 64 + kc + (hi << 4);
#pragma unroll
                for (int j = 0; j < 16; ++j) bm[j] = p[j];
            }
            acc = __builtin_amdgcn_wmma_f32_16x16x32_f16(false, a, false, bm, (short)0, acc, false, false);
        }
    }
    float* O = ker + (size_t)f * 32 * HWP;
#pragma unroll
    for (int v = 0; v < 8; ++v)
        O[(size_t)(mt * 16 + v + 8 * hi) * HWP + n] = acc[v];
}

// normalize 50-tap kernel: kern = ker - mean + 1/50
__global__ void k_kern(const float* ker, float* kern) {
    int idx = blockIdx.x * blockDim.x + threadIdx.x;
    if (idx >= 2 * BB * HWP) return;
    int p = idx & 4095, b = (idx >> 12) & 7, tensor = idx >> 15;
    float vals[50], s = 0.f;
#pragma unroll
    for (int t = 0; t < 2; ++t)
        for (int kk = 0; kk < 25; ++kk) {
            float v = ker[(((size_t)(tensor * 16 + b * 2 + t)) * 32 + kk) * HWP + p];
            vals[t * 25 + kk] = v; s += v;
        }
    float mean = s * (1.f / 50.f);
    float* dst = kern + ((size_t)(tensor * 8 + b)) * 50 * HWP + p;
#pragma unroll
    for (int i = 0; i < 50; ++i) dst[(size_t)i * HWP] = vals[i] - mean + 0.02f;
}

// apply per-pixel dynamic filter (edge padding)
__global__ void k_apply(const float* kreg, const float* vreg, const float* kern, float* kf, float* vf) {
    int idx = blockIdx.x * blockDim.x + threadIdx.x;
    if (idx >= 2 * BB * 64 * HWP) return;
    int p = idx & 4095, c = (idx >> 12) & 63, b = (idx >> 18) & 7, tensor = idx >> 21;
    int py = p >> 6, px = p & 63;
    const float* src = (tensor ? vreg : kreg) + (((size_t)b * 64 + c) * 2) * HWP;
    const float* kr = kern + ((size_t)(tensor * 8 + b)) * 50 * HWP + p;
    float acc = 0.f;
    for (int ti = 0; ti < 2; ++ti) {
        const float* st = src + (size_t)ti * HWP;
#pragma unroll
        for (int kk = 0; kk < 25; ++kk) {
            int yy = py + kk / 5 - 2, xx = px + kk % 5 - 2;
            yy = yy < 0 ? 0 : (yy > 63 ? 63 : yy);
            xx = xx < 0 ? 0 : (xx > 63 ? 63 : xx);
            acc += st[yy * 64 + xx] * kr[(size_t)(ti * 25 + kk) * HWP];
        }
    }
    (tensor ? vf : kf)[((size_t)b * 64 + c) * HWP + p] = acc;
}

// ---------------- attention ----------------

// pack q/k (L2-normalized, temp folded into q) and v to f16; qh/kh: [prob][s][16], vhT: [prob][16][1024]
__global__ void k_attn_prep(const float* qkv, const float* kf, const float* vf, const float* temp,
                            _Float16* qh, _Float16* kh, _Float16* vhT) {
    int idx = blockIdx.x * blockDim.x + threadIdx.x;
    if (idx >= 3 * 32 * 4 * 1024) return;
    int s = idx & 1023, hd = (idx >> 10) & 3, n = (idx >> 12) & 31, which = idx >> 17;
    int b = n >> 2, win = n & 3;
    int pix = ((win >> 1) * 32 + (s >> 5)) * 64 + (win & 1) * 32 + (s & 31);
    const float* src;
    if (which == 0)      src = qkv + ((size_t)b * 192 + hd * 16) * HWP + pix;
    else if (which == 1) src = kf + ((size_t)b * 64 + hd * 16) * HWP + pix;
    else                 src = vf + ((size_t)b * 64 + hd * 16) * HWP + pix;
    float v[16], ss = 0.f;
#pragma unroll
    for (int d = 0; d < 16; ++d) { v[d] = src[(size_t)d * HWP]; ss += v[d] * v[d]; }
    if (which < 2) {
        float inv = 1.f / fmaxf(sqrtf(ss), 1e-12f);
#pragma unroll
        for (int d = 0; d < 16; ++d) v[d] *= inv;
    }
    if (which == 0) {
        float t = temp[hd];
#pragma unroll
        for (int d = 0; d < 16; ++d) v[d] *= t;
    }
    int prob = n * 4 + hd;
    if (which < 2) {
        _Float16* dst = (which ? kh : qh) + ((size_t)prob * 1024 + s) * 16;
#pragma unroll
        for (int d = 0; d < 16; ++d) dst[d] = (_Float16)v[d];
    } else {
        _Float16* dst = vhT + (size_t)prob * 16 * 1024;
#pragma unroll
        for (int d = 0; d < 16; ++d) dst[(size_t)d * 1024 + s] = (_Float16)v[d];
    }
}

// flash attention: one wave = 16 query rows of one (window, head) problem
__global__ void __launch_bounds__(256) k_attn(const _Float16* qh, const _Float16* kh,
                                              const _Float16* vhT, float* attn_out) {
    __shared__ float    sS[8][16][32];
    __shared__ _Float16 sP[8][16][32];
    __shared__ float    sScale[8][16];
    int wv = threadIdx.x >> 5, lane = threadIdx.x & 31;
    int wave = blockIdx.x * 8 + wv;
    int qt = wave & 63, hd = (wave >> 6) & 3, n = wave >> 8;
    int prob = n * 4 + hd;
    const _Float16* Q = qh + (size_t)prob * 1024 * 16;
    const _Float16* K = kh + (size_t)prob * 1024 * 16;
    const _Float16* VT = vhT + (size_t)prob * 16 * 1024;
    int mrow = lane & 15, hi = lane >> 4;

    v16h aQ = vz16();
    {
        const _Float16* qp = Q + (size_t)(qt * 16 + mrow) * 16;
#pragma unroll
        for (int j = 0; j < 16; ++j) if (hi == 0) aQ[j] = qp[j];   // D padded 16->32 with zeros
    }
    v8f o = {};
    float mrun = -1e30f, lrun = 0.f;  // valid on lanes 0..15 (row = lane)

    for (int kc = 0; kc < 32; ++kc) {
#pragma unroll
        for (int sub = 0; sub < 2; ++sub) {
            v16h bK = vz16();
            if (hi == 0) {
                const _Float16* kp = K + (size_t)(kc * 32 + sub * 16 + mrow) * 16;
#pragma unroll
                for (int j = 0; j < 16; ++j) bK[j] = kp[j];
            }
            v8f c = {};
            c = __builtin_amdgcn_wmma_f32_16x16x32_f16(false, aQ, false, bK, (short)0, c, false, false);
#pragma unroll
            for (int v = 0; v < 8; ++v) sS[wv][v + 8 * hi][(lane & 15) + 16 * sub] = c[v];
        }
        asm volatile("s_wait_dscnt 0x0" ::: "memory");
        if (hi == 0) {
            float* row = &sS[wv][mrow][0];
            float cmax = -1e30f;
#pragma unroll
            for (int j = 0; j < 32; ++j) cmax = fmaxf(cmax, row[j]);
            float mnew = fmaxf(mrun, cmax);
            float sc = __expf(mrun - mnew), ssum = 0.f;
#pragma unroll
            for (int j = 0; j < 32; ++j) {
                float p = __expf(row[j] - mnew);
                ssum += p;
                sP[wv][mrow][j] = (_Float16)p;
            }
            lrun = lrun * sc + ssum;
            mrun = mnew;
            sScale[wv][mrow] = sc;
        }
        asm volatile("s_wait_dscnt 0x0" ::: "memory");
#pragma unroll
        for (int v = 0; v < 8; ++v) o[v] *= sScale[wv][v + 8 * hi];
        v16h aP;
        {
            const _Float16* pp = &sP[wv][mrow][0];
            int off = hi * 16;
#pragma unroll
            for (int j = 0; j < 8; ++j) aP[j] = pp[off + j];
#pragma unroll
            for (int j = 0; j < 8; ++j) aP[8 + j] = pp[8 + off + j];
        }
        v16h bV;
        {
            const _Float16* vp = VT + (size_t)(lane & 15) * 1024 + kc * 32 + hi * 16;
#pragma unroll
            for (int j = 0; j < 16; ++j) bV[j] = vp[j];
        }
        o = __builtin_amdgcn_wmma_f32_16x16x32_f16(false, aP, false, bV, (short)0, o, false, false);
    }
    if (hi == 0) sScale[wv][mrow] = 1.f / lrun;
    asm volatile("s_wait_dscnt 0x0" ::: "memory");
#pragma unroll
    for (int v = 0; v < 8; ++v) o[v] *= sScale[wv][v + 8 * hi];

    int b = n >> 2, win = n & 3, top = (win >> 1) * 32, left = (win & 1) * 32, d = lane & 15;
    float* O = attn_out + ((size_t)b * 64 + hd * 16 + d) * HWP;
#pragma unroll
    for (int v = 0; v < 8; ++v) {
        int s = qt * 16 + v + 8 * hi;
        O[(top + (s >> 5)) * 64 + left + (s & 31)] = o[v];
    }
}

// attn_out -> f16 transposed with nan_to_num(1e-5)
__global__ void k_ao_f16(const float* ao, _Float16* aoT) {
    int idx = blockIdx.x * blockDim.x + threadIdx.x;
    if (idx >= BB * HWP * 64) return;
    int c = idx & 63, p = (idx >> 6) & 4095, b = idx >> 18;
    float v = ao[((size_t)b * 64 + c) * HWP + p];
    if (!(fabsf(v) < 1e30f)) v = 1e-5f;   // NaN / +-Inf -> 1e-5
    aoT[idx] = (_Float16)v;
}

__global__ void k_gemm_proj(const _Float16* W, const _Float16* aoT, float* out) {
    int wave = (blockIdx.x * blockDim.x + threadIdx.x) >> 5;
    int lane = threadIdx.x & 31;
    int nt = wave % 256, mt = (wave / 256) & 3, b = wave / (256 * 4);
    const _Float16* BT = aoT + (size_t)b * HWP * 64;
    v8f acc = {};
#pragma unroll
    for (int k0 = 0; k0 < 64; k0 += 32) {
        v16h a = wmma_a_rm(W, 64, mt * 16, k0, lane);
        v16h bm = wmma_b_km(BT, 64, nt * 16, k0, lane);
        acc = __builtin_amdgcn_wmma_f32_16x16x32_f16(false, a, false, bm, (short)0, acc, false, false);
    }
    float* O = out + (size_t)b * 64 * HWP;
    int hi = lane >> 4, nn = nt * 16 + (lane & 15);
#pragma unroll
    for (int v = 0; v < 8; ++v)
        O[(size_t)(mt * 16 + v + 8 * hi) * HWP + nn] = acc[v];
}

// ---------------- launcher ----------------

extern "C" void kernel_launch(void* const* d_in, const int* in_sizes, int n_in,
                              void* d_out, int out_size, void* d_ws, size_t ws_size,
                              hipStream_t stream) {
    const float* x        = (const float*)d_in[0];
    const float* k_cached = (const float*)d_in[1];
    const float* v_cached = (const float*)d_in[2];
    const float* W_qkv    = (const float*)d_in[3];
    const float* W_dw     = (const float*)d_in[4];
    const float* W_f1     = (const float*)d_in[5];
    const float* W_f2     = (const float*)d_in[6];
    const float* W_proj   = (const float*)d_in[7];
    const float* temp     = (const float*)d_in[8];

    float* out0 = (float*)d_out;
    float* outk = out0 + NOUT0;
    float* outv = outk + NKV;

    char* w = (char*)d_ws;
    size_t off = 0;
    auto alloc = [&](size_t bytes) -> void* {
        void* p = w + off;
        off += (bytes + 255) & ~(size_t)255;
        return p;
    };
    _Float16* xT     = (_Float16*)alloc((size_t)BB * HWP * 64 * 2);
    _Float16* WqkvH  = (_Float16*)alloc(12288 * 2);
    _Float16* Wf1H   = (_Float16*)alloc(36864 * 2);
    _Float16* Wf2H   = (_Float16*)alloc(18432 * 2);
    _Float16* WprojH = (_Float16*)alloc(4096 * 2);
    float*    qkv0   = (float*)alloc((size_t)BB * 192 * HWP * 4);
    float*    qkv    = (float*)alloc((size_t)BB * 192 * HWP * 4);
    _Float16* yT     = (_Float16*)alloc((size_t)32 * HWP * 64 * 2);
    _Float16* fT     = (_Float16*)alloc((size_t)32 * HWP * 64 * 2);
    float*    ker    = (float*)alloc((size_t)32 * 32 * HWP * 4);
    float*    kern   = (float*)alloc((size_t)2 * BB * 50 * HWP * 4);
    float*    kf     = (float*)alloc((size_t)BB * 64 * HWP * 4);
    float*    vf     = (float*)alloc((size_t)BB * 64 * HWP * 4);
    _Float16* qh     = (_Float16*)alloc((size_t)32 * 4 * 1024 * 16 * 2);
    _Float16* kh     = (_Float16*)alloc((size_t)32 * 4 * 1024 * 16 * 2);
    _Float16* vhT    = (_Float16*)alloc((size_t)32 * 4 * 16 * 1024 * 2);
    float*    aout   = (float*)alloc((size_t)BB * 64 * HWP * 4);
    _Float16* aoT    = (_Float16*)alloc((size_t)BB * HWP * 64 * 2);

    k_prep_w<<<(71680 + 255) / 256, 256, 0, stream>>>(W_qkv, W_f1, W_f2, W_proj, WqkvH, Wf1H, Wf2H, WprojH);
    k_x_f16<<<(BB * HWP * 64) / 256, 256, 0, stream>>>(x, xT);
    k_gemm_qkv<<<(24576 * 32) / 256, 256, 0, stream>>>(WqkvH, xT, qkv0);
    k_depthwise<<<(BB * 192 * HWP) / 256, 256, 0, stream>>>(qkv0, W_dw, qkv);
    k_concat<<<(BB * 64 * HWP) / 256, 256, 0, stream>>>(k_cached, v_cached, qkv, outk, outv);
    k_frames_f16<<<(32 * HWP * 64) / 256, 256, 0, stream>>>(outk, outv, yT);
    k_conv1<<<(32768 * 32) / 256, 256, 0, stream>>>(Wf1H, yT, fT);
    k_conv2<<<(16384 * 32) / 256, 256, 0, stream>>>(Wf2H, fT, ker);
    k_kern<<<(2 * BB * HWP) / 256, 256, 0, stream>>>(ker, kern);
    k_apply<<<(2 * BB * 64 * HWP) / 256, 256, 0, stream>>>(outk, outv, kern, kf, vf);
    k_attn_prep<<<(3 * 32 * 4 * 1024) / 256, 256, 0, stream>>>(qkv, kf, vf, temp, qh, kh, vhT);
    k_attn<<<8192 / 8, 256, 0, stream>>>(qh, kh, vhT, aout);
    k_ao_f16<<<(BB * HWP * 64) / 256, 256, 0, stream>>>(aout, aoT);
    k_gemm_proj<<<(8192 * 32) / 256, 256, 0, stream>>>(WprojH, aoT, out0);
}
